// ContrastiveLoss_44994077393418
// MI455X (gfx1250) — compile-verified, matrix operand order checked by reading, not run on hardware
//
#include <hip/hip_runtime.h>

#define BATCH    4096
#define DIM      512
#define NROW     8192            // 2*BATCH
#define INV_TEMP 10.0f
#define COLSPLIT 16
#define ROWS_PER_BLOCK 256       // 8 waves * 32 rows
#define NBLK_ROW (NROW / ROWS_PER_BLOCK)   // 32

typedef __attribute__((ext_vector_type(16))) __bf16 v16bf;
typedef __attribute__((ext_vector_type(8)))  float  v8f;

__device__ __forceinline__ unsigned short f2bf_rne(float f) {
  unsigned int u = __float_as_uint(f);
  u += 0x7FFFu + ((u >> 16) & 1u);          // round-to-nearest-even
  return (unsigned short)(u >> 16);
}
__device__ __forceinline__ float bf2f(unsigned short h) {
  return __uint_as_float(((unsigned int)h) << 16);
}

// Build one 16x32 bf16 operand fragment (A or B) for the wave.
// p points at row_base + k, where row_base already includes the +8*(lane>=16)
// half-wave K offset. halves[0..7] = K {0..7}, halves[8..15] = K {16..23}
// relative to p — matching the documented CDNA5 16-bit A/B VGPR layout.
__device__ __forceinline__ v16bf load_frag(const unsigned short* p) {
  union { uint4 u[2]; v16bf v; } r;
  r.u[0] = *reinterpret_cast<const uint4*>(p);       // 8 bf16, 16B aligned
  r.u[1] = *reinterpret_cast<const uint4*>(p + 16);  // 8 bf16 at +32B
  return r.v;
}

__device__ __forceinline__ v8f wmma_bf16(v16bf a, v16bf b, v8f c) {
  return __builtin_amdgcn_wmma_f32_16x16x32_bf16(false, a, false, b,
                                                 (short)0, c, false, false);
}

// ---------------------------------------------------------------- zero
__global__ void nt_zero(float* __restrict__ srow, float* __restrict__ out) {
  int i = blockIdx.x * blockDim.x + threadIdx.x;
  if (i < NROW) srow[i] = 0.0f;
  if (i == 0) out[0] = 0.0f;
}

// ------------------------------------------------------------ normalize
// One block (128 threads) per pair index i: normalizes proj_1[i] and
// proj_2[i], emits bf16 hi/lo splits for rows i and BATCH+i, and the
// positive logit pos[i] = dot(z1,z2)/T computed in fp32.
__global__ __launch_bounds__(128)
void nt_normalize(const float* __restrict__ p1, const float* __restrict__ p2,
                  unsigned short* __restrict__ zhi, unsigned short* __restrict__ zlo,
                  float* __restrict__ pos) {
  const int row = blockIdx.x;         // 0..BATCH-1
  const int tid = threadIdx.x;        // 0..127
  const float* a = p1 + (size_t)row * DIM;
  const float* b = p2 + (size_t)row * DIM;

  float x1[4], x2[4];
  float ss1 = 0.0f, ss2 = 0.0f, dd = 0.0f;
#pragma unroll
  for (int i = 0; i < 4; ++i) {
    int j = tid + 128 * i;
    x1[i] = a[j]; x2[i] = b[j];
    ss1 = fmaf(x1[i], x1[i], ss1);
    ss2 = fmaf(x2[i], x2[i], ss2);
    dd  = fmaf(x1[i], x2[i], dd);
  }
#pragma unroll
  for (int m = 16; m >= 1; m >>= 1) {
    ss1 += __shfl_xor(ss1, m, 32);
    ss2 += __shfl_xor(ss2, m, 32);
    dd  += __shfl_xor(dd,  m, 32);
  }
  __shared__ float red[3][4];
  const int wv = tid >> 5;
  if ((tid & 31) == 0) { red[0][wv] = ss1; red[1][wv] = ss2; red[2][wv] = dd; }
  __syncthreads();
  ss1 = red[0][0] + red[0][1] + red[0][2] + red[0][3];
  ss2 = red[1][0] + red[1][1] + red[1][2] + red[1][3];
  dd  = red[2][0] + red[2][1] + red[2][2] + red[2][3];

  const float inv1 = 1.0f / fmaxf(sqrtf(ss1), 1e-12f);
  const float inv2 = 1.0f / fmaxf(sqrtf(ss2), 1e-12f);
  if (tid == 0) pos[row] = dd * inv1 * inv2 * INV_TEMP;

  unsigned short* h1p = zhi + (size_t)row * DIM;
  unsigned short* l1p = zlo + (size_t)row * DIM;
  unsigned short* h2p = zhi + (size_t)(BATCH + row) * DIM;
  unsigned short* l2p = zlo + (size_t)(BATCH + row) * DIM;
#pragma unroll
  for (int i = 0; i < 4; ++i) {
    int j = tid + 128 * i;
    float z1 = x1[i] * inv1;
    float z2 = x2[i] * inv2;
    unsigned short h1 = f2bf_rne(z1);
    unsigned short h2 = f2bf_rne(z2);
    h1p[j] = h1;  l1p[j] = f2bf_rne(z1 - bf2f(h1));
    h2p[j] = h2;  l2p[j] = f2bf_rne(z2 - bf2f(h2));
  }
}

// ----------------------------------------------------- GEMM + streaming Σexp
// Each wave owns a 32x32 output block (2x2 WMMA tiles): 16 b128 loads feed
// 12 WMMAs per k-step. Fixed-shift Σexp partials are additive, so column
// splits combine via a single atomicAdd per row per wave.
__global__ __launch_bounds__(256)
void nt_gemm_lse(const unsigned short* __restrict__ zhi,
                 const unsigned short* __restrict__ zlo,
                 float* __restrict__ srow) {
  const int wave = threadIdx.x >> 5;
  const int lane = threadIdx.x & 31;
  const int half = lane >> 4;           // K half-offset selector
  const int l16  = lane & 15;

  const int rowBlock = blockIdx.x & (NBLK_ROW - 1);
  const int colSplit = blockIdx.x / NBLK_ROW;
  const int m0 = rowBlock * ROWS_PER_BLOCK + wave * 32;

  const unsigned short* Ahi0 = zhi + (size_t)(m0 + l16) * DIM + 8 * half;
  const unsigned short* Alo0 = zlo + (size_t)(m0 + l16) * DIM + 8 * half;
  const unsigned short* Ahi1 = Ahi0 + (size_t)16 * DIM;
  const unsigned short* Alo1 = Alo0 + (size_t)16 * DIM;

  float s[2][8];
#pragma unroll
  for (int mi = 0; mi < 2; ++mi)
#pragma unroll
    for (int v = 0; v < 8; ++v) s[mi][v] = 0.0f;

  const int nCols   = NROW / COLSPLIT;        // 512
  const int n_begin = colSplit * nCols;
  const int nTiles  = nCols / 32;             // 16

  for (int tn = 0; tn < nTiles; ++tn) {
    const int n0    = n_begin + tn * 32;
    const int ncol0 = n0 + l16;
    const unsigned short* Bhi0 = zhi + (size_t)ncol0 * DIM + 8 * half;
    const unsigned short* Blo0 = zlo + (size_t)ncol0 * DIM + 8 * half;
    const unsigned short* Bhi1 = Bhi0 + (size_t)16 * DIM;
    const unsigned short* Blo1 = Blo0 + (size_t)16 * DIM;

    // Hide L2 latency on the next column tile (global_prefetch_b8).
    if (tn + 1 < nTiles) {
      __builtin_prefetch(Bhi0 + (size_t)32 * DIM, 0, 0);
      __builtin_prefetch(Blo0 + (size_t)32 * DIM, 0, 0);
    }

    v8f c00 = {}; v8f c01 = {}; v8f c10 = {}; v8f c11 = {};
#pragma unroll 2
    for (int k = 0; k < DIM; k += 32) {
      v16bf ah0 = load_frag(Ahi0 + k);
      v16bf al0 = load_frag(Alo0 + k);
      v16bf ah1 = load_frag(Ahi1 + k);
      v16bf al1 = load_frag(Alo1 + k);
      v16bf bh0 = load_frag(Bhi0 + k);
      v16bf bl0 = load_frag(Blo0 + k);
      v16bf bh1 = load_frag(Bhi1 + k);
      v16bf bl1 = load_frag(Blo1 + k);
      // hi*hi products (4 independent chains)
      c00 = wmma_bf16(ah0, bh0, c00);
      c01 = wmma_bf16(ah0, bh1, c01);
      c10 = wmma_bf16(ah1, bh0, c10);
      c11 = wmma_bf16(ah1, bh1, c11);
      // hi*lo correction
      c00 = wmma_bf16(ah0, bl0, c00);
      c01 = wmma_bf16(ah0, bl1, c01);
      c10 = wmma_bf16(ah1, bl0, c10);
      c11 = wmma_bf16(ah1, bl1, c11);
      // lo*hi correction
      c00 = wmma_bf16(al0, bh0, c00);
      c01 = wmma_bf16(al0, bh1, c01);
      c10 = wmma_bf16(al1, bh0, c10);
      c11 = wmma_bf16(al1, bh1, c11);
    }
    // Epilogue: fixed-shift sum of exp, diagonal masked.
#pragma unroll
    for (int v = 0; v < 8; ++v) {
      const int grow0 = m0 + v + 8 * half;        // rows of tile mi=0
      const int grow1 = grow0 + 16;               // rows of tile mi=1
      const int gcol0 = ncol0;                    // cols of tile ni=0
      const int gcol1 = ncol0 + 16;               // cols of tile ni=1
      float x00 = c00[v] * INV_TEMP - INV_TEMP;   // in [-20, 0]
      float x01 = c01[v] * INV_TEMP - INV_TEMP;
      float x10 = c10[v] * INV_TEMP - INV_TEMP;
      float x11 = c11[v] * INV_TEMP - INV_TEMP;
      s[0][v] += ((gcol0 == grow0) ? 0.0f : __expf(x00))
               + ((gcol1 == grow0) ? 0.0f : __expf(x01));
      s[1][v] += ((gcol0 == grow1) ? 0.0f : __expf(x10))
               + ((gcol1 == grow1) ? 0.0f : __expf(x11));
    }
  }
  // Reduce each row's 16 column-lanes, one atomic per row.
#pragma unroll
  for (int mi = 0; mi < 2; ++mi) {
#pragma unroll
    for (int v = 0; v < 8; ++v) {
      float t = s[mi][v];
      t += __shfl_xor(t, 1, 32);
      t += __shfl_xor(t, 2, 32);
      t += __shfl_xor(t, 4, 32);
      t += __shfl_xor(t, 8, 32);
      if (l16 == 0) atomicAdd(&srow[m0 + mi * 16 + v + 8 * half], t);
    }
  }
}

// ---------------------------------------------------------------- finalize
__global__ __launch_bounds__(256)
void nt_finalize(const float* __restrict__ srow, const float* __restrict__ pos,
                 float* __restrict__ out) {
  const int r = blockIdx.x * blockDim.x + threadIdx.x;
  float c = 0.0f;
  if (r < NROW) {
    float lse = INV_TEMP + logf(srow[r]);
    c = (lse - pos[r & (BATCH - 1)]) * (1.0f / (float)NROW);
  }
#pragma unroll
  for (int m = 16; m >= 1; m >>= 1) c += __shfl_xor(c, m, 32);
  __shared__ float red[8];
  const int wv = threadIdx.x >> 5;
  if ((threadIdx.x & 31) == 0) red[wv] = c;
  __syncthreads();
  if (threadIdx.x == 0) {
    float t = 0.0f;
#pragma unroll
    for (int i = 0; i < 8; ++i) t += red[i];
    atomicAdd(out, t);
  }
}

extern "C" void kernel_launch(void* const* d_in, const int* in_sizes, int n_in,
                              void* d_out, int out_size, void* d_ws, size_t ws_size,
                              hipStream_t stream) {
  (void)in_sizes; (void)n_in; (void)out_size; (void)ws_size;
  const float* p1 = (const float*)d_in[0];
  const float* p2 = (const float*)d_in[1];
  float* out = (float*)d_out;

  // Workspace layout (all 16B-aligned offsets):
  char* ws = (char*)d_ws;
  unsigned short* zhi  = (unsigned short*)(ws);                              // 8 MB
  unsigned short* zlo  = (unsigned short*)(ws + (size_t)NROW * DIM * 2);     // 8 MB
  float*          pos  = (float*)(ws + (size_t)NROW * DIM * 4);              // 16 KB
  float*          srow = (float*)(ws + (size_t)NROW * DIM * 4 + BATCH * 4);  // 32 KB

  nt_zero     <<<NROW / 256, 256, 0, stream>>>(srow, out);
  nt_normalize<<<BATCH, 128, 0, stream>>>(p1, p2, zhi, zlo, pos);
  nt_gemm_lse <<<NBLK_ROW * COLSPLIT, 256, 0, stream>>>(zhi, zlo, srow);
  nt_finalize <<<NROW / 256, 256, 0, stream>>>(srow, pos, out);
}
